// self_attention_all_7834020348653
// MI455X (gfx1250) — matrix operand order repaired and checked by reading, not compiled, not run on hardware
//
#include <hip/hip_runtime.h>

typedef __attribute__((ext_vector_type(16))) __bf16 v16bf;
typedef __attribute__((ext_vector_type(8)))  __bf16 v8bf;
typedef __attribute__((ext_vector_type(8)))  float  v8f;

constexpr int Bb  = 8;
constexpr int Nn  = 2048;
constexpr int Cin = 256;
constexpr int C   = 256;
constexpr int C3  = 768;
constexpr int H   = 8;
constexpr int D   = 32;

__device__ __forceinline__ __bf16 f2bf(float x) {
    unsigned u = __builtin_bit_cast(unsigned, x);
    u += 0x7FFFu + ((u >> 16) & 1u);          // round-to-nearest-even
    unsigned short s = (unsigned short)(u >> 16);
    return __builtin_bit_cast(__bf16, s);
}

__device__ __forceinline__ v8f vzero8() {
    v8f z;
#pragma unroll
    for (int i = 0; i < 8; ++i) z[i] = 0.0f;
    return z;
}

__device__ __forceinline__ v16bf cat8(v8bf lo, v8bf hi) {
    return __builtin_shufflevector(lo, hi, 0,1,2,3,4,5,6,7,8,9,10,11,12,13,14,15);
}

__device__ __forceinline__ v8f wmma_bf16(v16bf a, v16bf b, v8f c) {
    // D = A(16x32 bf16) * B(32x16 bf16) + C(16x16 f32)
    return __builtin_amdgcn_wmma_f32_16x16x32_bf16(
        /*neg_a=*/false, a, /*neg_b=*/false, b,
        /*c_mod=*/(short)0, c, /*reuse_a=*/false, /*reuse_b=*/false);
}

// Async global -> LDS copy, 16 bytes per lane, tracked by ASYNCcnt.
// lds_off: byte address in LDS (low 32 bits of a generic shared pointer),
// gptr: 64-bit global address in a VGPR pair.
__device__ __forceinline__ void async_copy_b128(unsigned lds_off, const void* gptr) {
    asm volatile("global_load_async_to_lds_b128 %0, %1, off"
                 :: "v"(lds_off), "v"(gptr) : "memory");
}

__device__ __forceinline__ void wait_asynccnt0() {
    asm volatile("s_wait_asynccnt 0x0" ::: "memory");
}

// ---------------------------------------------------------------------------
// Kernel 1: QKV projection.  [16384,256]f32 @ [256,768]f32 -> scatter bf16
// into Q/K/V buffers laid out [B,H,N,D].  Q is pre-scaled by C^-0.5 = 1/16.
// Block: 256 thr (8 waves). Tile: 128 rows x 64 cols; wave = 16-row strip.
// ---------------------------------------------------------------------------
__global__ __launch_bounds__(256) void qkv_kernel(
    const float* __restrict__ in, const float* __restrict__ W,
    __bf16* __restrict__ qh, __bf16* __restrict__ kh, __bf16* __restrict__ vh)
{
    __shared__ __bf16 BsT[64][32];   // W tile, transposed (col-major) for b-frags

    const int tid   = threadIdx.x;
    const int wave  = tid >> 5;
    const int lane  = tid & 31;
    const int ln    = lane & 15;
    const int khalf = lane >> 4;

    const int mbase = blockIdx.x * 128;
    const int cbase = blockIdx.y * 64;
    const int lrow  = mbase + wave * 16 + ln;

    v8f acc[4];
#pragma unroll
    for (int ct = 0; ct < 4; ++ct) acc[ct] = vzero8();

    for (int k0 = 0; k0 < Cin; k0 += 32) {
        __syncthreads();
        for (int e = tid; e < 32 * 64; e += 256) {
            int kk = e >> 6, cc = e & 63;
            BsT[cc][kk] = f2bf(W[(k0 + kk) * C3 + cbase + cc]);
        }
        __syncthreads();

        // A fragment straight from global (f32 -> bf16 in regs)
        const float* ar = in + (size_t)lrow * Cin + k0 + khalf * 16;
        v16bf a;
#pragma unroll
        for (int j = 0; j < 8; ++j) {
            a[j]     = f2bf(ar[j]);        // K = khalf*16 + j
            a[j + 8] = f2bf(ar[8 + j]);    // K = khalf*16 + 8 + j
        }
#pragma unroll
        for (int ct = 0; ct < 4; ++ct) {
            v16bf b = *reinterpret_cast<const v16bf*>(&BsT[ct * 16 + ln][khalf * 16]);
            acc[ct] = wmma_bf16(a, b, acc[ct]);
        }
    }

    // D layout: acc[ct][r] holds (M = r + khalf*8, Ncol = ln)
#pragma unroll
    for (int ct = 0; ct < 4; ++ct) {
#pragma unroll
        for (int r = 0; r < 8; ++r) {
            int m = mbase + wave * 16 + r + khalf * 8;
            int c = cbase + ct * 16 + ln;
            int b = m >> 11, n = m & (Nn - 1);
            int t = c >> 8, hh = (c >> 5) & 7, dd = c & 31;
            float v = acc[ct][r] * ((t == 0) ? 0.0625f : 1.0f);
            __bf16* dst = (t == 0) ? qh : (t == 1) ? kh : vh;
            dst[(((size_t)b * H + hh) * Nn + n) * D + dd] = f2bf(v);
        }
    }
}

// ---------------------------------------------------------------------------
// Kernel 2: flash attention per (b,h).  Block: 256 thr, 128 query rows,
// each wave owns a 16-row strip.  64-key K/V tiles are double-buffered in
// LDS via async global->LDS copies (ASYNCcnt), overlapping copy with WMMA.
// ---------------------------------------------------------------------------
__global__ __launch_bounds__(256) void attn_kernel(
    const __bf16* __restrict__ qh, const __bf16* __restrict__ kh,
    const __bf16* __restrict__ vh, __bf16* __restrict__ xout)
{
    __shared__ __bf16 Ks[2][64][32];   // [buf][key][d]
    __shared__ __bf16 Vs[2][64][32];   // [buf][key][d]
    __shared__ __bf16 Ps[8][16][64];   // per-wave P relayout slab

    const int tid   = threadIdx.x;
    const int wave  = tid >> 5;
    const int lane  = tid & 31;
    const int ln    = lane & 15;
    const int khalf = lane >> 4;

    const int bh    = blockIdx.y;                 // b*H + h
    const int qbase = blockIdx.x * 128 + wave * 16;

    const __bf16* Q = qh + (size_t)bh * Nn * D;
    const __bf16* K = kh + (size_t)bh * Nn * D;
    const __bf16* V = vh + (size_t)bh * Nn * D;

    const unsigned ksBase = (unsigned)(size_t)(&Ks[0][0][0]);
    const unsigned vsBase = (unsigned)(size_t)(&Vs[0][0][0]);

    // Each 64-key tile is 64*32 bf16 = 4096 contiguous bytes in [B,H,N,D]
    // layout: one b128 async copy per thread covers it exactly.
    auto issue_tile = [&](int kv, int buf) {
        const char* kg = (const char*)(K + (size_t)kv * D) + tid * 16;
        const char* vg = (const char*)(V + (size_t)kv * D) + tid * 16;
        async_copy_b128(ksBase + (unsigned)(buf * 4096 + tid * 16), kg);
        async_copy_b128(vsBase + (unsigned)(buf * 4096 + tid * 16), vg);
    };

    // Resident Q a-fragment (d == 32 == one bf16 WMMA K-step)
    v16bf aq;
    {
        const __bf16* qr = Q + (size_t)(qbase + ln) * D + khalf * 16;
        v8bf lo = *reinterpret_cast<const v8bf*>(qr);
        v8bf hi = *reinterpret_cast<const v8bf*>(qr + 8);
        aq = cat8(lo, hi);
    }

    float m[8], l[8];
    v8f O[2];
#pragma unroll
    for (int r = 0; r < 8; ++r) { m[r] = -3.0e38f; l[r] = 0.0f; }
    O[0] = vzero8(); O[1] = vzero8();

    // Prologue: fetch tile 0
    issue_tile(0, 0);
    wait_asynccnt0();
    __syncthreads();

    for (int kv = 0; kv < Nn; kv += 64) {
        const int buf = (kv >> 6) & 1;
        if (kv + 64 < Nn) issue_tile(kv + 64, buf ^ 1);   // overlap with compute

        // S = Q K^T  (4 key sub-tiles of 16)
        v8f S[4];
#pragma unroll
        for (int kt = 0; kt < 4; ++kt) {
            v16bf bk = *reinterpret_cast<const v16bf*>(&Ks[buf][kt * 16 + ln][khalf * 16]);
            S[kt] = wmma_bf16(aq, bk, vzero8());
        }

        // Online softmax (rows live as r + khalf*8; reductions stay in 16-lane halves)
#pragma unroll
        for (int r = 0; r < 8; ++r) {
            float rm = fmaxf(fmaxf(S[0][r], S[1][r]), fmaxf(S[2][r], S[3][r]));
#pragma unroll
            for (int off = 1; off < 16; off <<= 1)
                rm = fmaxf(rm, __shfl_xor(rm, off, 32));
            float mn    = fmaxf(m[r], rm);
            float alpha = __expf(m[r] - mn);
            m[r] = mn;
            float rs = 0.0f;
#pragma unroll
            for (int kt = 0; kt < 4; ++kt) {
                float e_ = __expf(S[kt][r] - mn);
                S[kt][r] = e_;                       // becomes P
                rs += e_;
            }
#pragma unroll
            for (int off = 1; off < 16; off <<= 1)
                rs += __shfl_xor(rs, off, 32);
            l[r] = l[r] * alpha + rs;
            O[0][r] *= alpha;
            O[1][r] *= alpha;
        }

        // Re-layout P (D-format -> A-format) through per-wave LDS slab
#pragma unroll
        for (int kt = 0; kt < 4; ++kt)
#pragma unroll
            for (int r = 0; r < 8; ++r)
                Ps[wave][r + khalf * 8][kt * 16 + ln] = f2bf(S[kt][r]);

        // O += P @ V   (K = 64 keys -> 2 WMMA K-steps; N = 32 d -> 2 tiles)
#pragma unroll
        for (int ks = 0; ks < 2; ++ks) {
            const __bf16* pr = &Ps[wave][ln][ks * 32 + khalf * 16];
            v8bf plo = *reinterpret_cast<const v8bf*>(pr);
            v8bf phi = *reinterpret_cast<const v8bf*>(pr + 8);
            v16bf ap = cat8(plo, phi);
#pragma unroll
            for (int dt = 0; dt < 2; ++dt) {
                // B-frag element j = V[key = ks*32 + khalf*16 + j][d = dt*16 + ln]
                v16bf bv;
#pragma unroll
                for (int j = 0; j < 16; ++j)
                    bv[j] = Vs[buf][ks * 32 + khalf * 16 + j][dt * 16 + ln];
                O[dt] = wmma_bf16(ap, bv, O[dt]);
            }
        }

        // Next tile's async copy must have landed, and all waves must be done
        // reading this buffer, before it is overwritten next iteration.
        wait_asynccnt0();
        __syncthreads();
    }

    // Finalize and store with the reference's faithful [B,h,N,d]->[B,N,C] reshape
    const int b = bh >> 3, hh = bh & 7;
#pragma unroll
    for (int dt = 0; dt < 2; ++dt) {
#pragma unroll
        for (int r = 0; r < 8; ++r) {
            int nn = qbase + r + khalf * 8;
            int dd = dt * 16 + ln;
            float v = O[dt][r] / l[r];
            int nprime = hh * 256 + (nn >> 3);
            int cprime = ((nn & 7) << 5) + dd;
            xout[((size_t)b * Nn + nprime) * C + cprime] = f2bf(v);
        }
    }
}

// ---------------------------------------------------------------------------
// Kernel 3: output projection + bias + ReLU.  x(bf16)[16384,256] @ [256,256]
// ---------------------------------------------------------------------------
__global__ __launch_bounds__(256) void proj_kernel(
    const __bf16* __restrict__ x, const float* __restrict__ pw,
    const float* __restrict__ pb, float* __restrict__ out)
{
    __shared__ __bf16 BsT[64][32];

    const int tid   = threadIdx.x;
    const int wave  = tid >> 5;
    const int lane  = tid & 31;
    const int ln    = lane & 15;
    const int khalf = lane >> 4;

    const int mbase = blockIdx.x * 128;
    const int cbase = blockIdx.y * 64;
    const int lrow  = mbase + wave * 16 + ln;

    v8f acc[4];
#pragma unroll
    for (int ct = 0; ct < 4; ++ct) acc[ct] = vzero8();

    for (int k0 = 0; k0 < C; k0 += 32) {
        __syncthreads();
        for (int e = tid; e < 32 * 64; e += 256) {
            int kk = e >> 6, cc = e & 63;
            BsT[cc][kk] = f2bf(pw[(k0 + kk) * C + cbase + cc]);
        }
        __syncthreads();

        const __bf16* ar = x + (size_t)lrow * C + k0 + khalf * 16;
        v8bf lo = *reinterpret_cast<const v8bf*>(ar);
        v8bf hi = *reinterpret_cast<const v8bf*>(ar + 8);
        v16bf a = cat8(lo, hi);
#pragma unroll
        for (int ct = 0; ct < 4; ++ct) {
            v16bf b = *reinterpret_cast<const v16bf*>(&BsT[ct * 16 + ln][khalf * 16]);
            acc[ct] = wmma_bf16(a, b, acc[ct]);
        }
    }

#pragma unroll
    for (int ct = 0; ct < 4; ++ct) {
#pragma unroll
        for (int r = 0; r < 8; ++r) {
            int mrow = mbase + wave * 16 + r + khalf * 8;
            int c    = cbase + ct * 16 + ln;
            float v  = acc[ct][r] + pb[c];
            out[(size_t)mrow * C + c] = fmaxf(v, 0.0f);
        }
    }
}

// ---------------------------------------------------------------------------
extern "C" void kernel_launch(void* const* d_in, const int* in_sizes, int n_in,
                              void* d_out, int out_size, void* d_ws, size_t ws_size,
                              hipStream_t stream) {
    (void)in_sizes; (void)n_in; (void)out_size; (void)ws_size;

    const float* in  = (const float*)d_in[0];   // [8,2048,256]
    const float* W   = (const float*)d_in[1];   // [256,768]
    const float* pw  = (const float*)d_in[2];   // [256,256]
    const float* pb  = (const float*)d_in[3];   // [256]
    float*       out = (float*)d_out;           // [8,2048,256]

    const size_t QSZ = (size_t)Bb * H * Nn * D; // 4,194,304 elems
    __bf16* qh = (__bf16*)d_ws;
    __bf16* kh = qh + QSZ;
    __bf16* vh = kh + QSZ;
    __bf16* xb = vh + QSZ;                      // [8,2048,256] bf16

    qkv_kernel <<<dim3(128, 12), 256, 0, stream>>>(in, W, qh, kh, vh);
    attn_kernel<<<dim3(16,  64), 256, 0, stream>>>(qh, kh, vh, xb);
    proj_kernel<<<dim3(128,  4), 256, 0, stream>>>(xb, pw, pb, out);
}
